// DeepSpeedMoEWrapper_19439021982128
// MI455X (gfx1250) — matrix operand order repaired and checked by reading, not compiled, not run on hardware
//
#include <hip/hip_runtime.h>

#define E_ 128
#define D_ 1024
#define C_ 256
#define T_ 8192
#define NTILE 128
#define KTILE 32

typedef __attribute__((ext_vector_type(16))) __bf16 v16bf;
typedef __attribute__((ext_vector_type(8)))  float  v8f;

union FragBF { uint4 q[2]; v16bf v; };

// f32 -> bf16 round-to-nearest-even
static __device__ __forceinline__ unsigned short f2bf(float x) {
    unsigned u = __float_as_uint(x);
    unsigned r = u + 0x7FFFu + ((u >> 16) & 1u);
    return (unsigned short)(r >> 16);
}

// ---------------------------------------------------------------------------
// Kernel 0: zero the bf16 dispatch buffer and the per-expert slot counters.
// ---------------------------------------------------------------------------
__global__ void moe_zero(uint4* __restrict__ bufA, int* __restrict__ cnt) {
    size_t i = (size_t)blockIdx.x * blockDim.x + threadIdx.x;
    if (i < (size_t)(E_ * C_ * D_ / 8)) bufA[i] = make_uint4(0u, 0u, 0u, 0u);
    if (i < E_) cnt[i] = 0;
}

// ---------------------------------------------------------------------------
// Kernel 1: gating. One 128-thread block per token; thread e computes
// logits[t][e]; thread 0 finds top-2, normalized combine weights, and grabs
// per-expert slots with atomics (slot identity doesn't affect the output).
// ---------------------------------------------------------------------------
__global__ void moe_gate(const float* __restrict__ x, const float* __restrict__ wg,
                         int* __restrict__ cnt,
                         int* __restrict__ e1a, int* __restrict__ e2a,
                         int* __restrict__ s1a, int* __restrict__ s2a,
                         float* __restrict__ g1a, float* __restrict__ g2a) {
    __shared__ float xs[D_];
    __shared__ float lg[E_];
    int t = blockIdx.x;
    const float* xr = x + (size_t)t * D_;
    for (int i = threadIdx.x; i < D_; i += 128) xs[i] = xr[i];
    __syncthreads();
    int e = threadIdx.x;
    float acc = 0.f;
#pragma unroll 8
    for (int d = 0; d < D_; ++d) acc += xs[d] * wg[d * E_ + e];
    lg[e] = acc;
    __syncthreads();
    if (threadIdx.x == 0) {
        int e1 = 0; float l1 = lg[0];
        for (int i = 1; i < E_; ++i) { float v = lg[i]; if (v > l1) { l1 = v; e1 = i; } }
        int e2 = (e1 == 0) ? 1 : 0; float l2 = lg[e2];
        for (int i = 0; i < E_; ++i) {
            if (i == e1) continue;
            float v = lg[i]; if (v > l2) { l2 = v; e2 = i; }
        }
        // normalized top-2 softmax weights depend only on (l1, l2)
        float w2 = __expf(l2 - l1);
        float inv = 1.f / (1.f + w2);
        int s1 = atomicAdd(&cnt[e1], 1);
        int s2 = atomicAdd(&cnt[e2], 1);
        e1a[t] = e1; e2a[t] = e2; s1a[t] = s1; s2a[t] = s2;
        g1a[t] = inv; g2a[t] = w2 * inv;
    }
}

// ---------------------------------------------------------------------------
// Kernel 2: dispatch tokens (f32 -> bf16) into per-expert slot rows.
// ---------------------------------------------------------------------------
__global__ void moe_dispatch(const float* __restrict__ x,
                             const int* __restrict__ e1a, const int* __restrict__ e2a,
                             const int* __restrict__ s1a, const int* __restrict__ s2a,
                             unsigned short* __restrict__ bufA) {
    int t = blockIdx.x;
    int i = threadIdx.x;                       // each thread: 4 elements
    float4 xv = ((const float4*)(x + (size_t)t * D_))[i];
    ushort4 bv;
    bv.x = f2bf(xv.x); bv.y = f2bf(xv.y); bv.z = f2bf(xv.z); bv.w = f2bf(xv.w);
    int e1 = e1a[t], s1 = s1a[t];
    if (s1 < C_) ((ushort4*)(bufA + (size_t)(e1 * C_ + s1) * D_))[i] = bv;
    int e2 = e2a[t], s2 = s2a[t];
    if (s2 < C_) ((ushort4*)(bufA + (size_t)(e2 * C_ + s2) * D_))[i] = bv;
}

// ---------------------------------------------------------------------------
// Kernel 3: grouped expert GEMM, bf16 WMMA, f32 accumulation.
// Grid (D/NTILE, E); 512 threads = 16 waves; wave w owns output rows
// [16w,16w+16) x NTILE. Full M=256 per block => each fp32 weight element is
// read from HBM exactly once (512 MB floor at 23.3 TB/s).
// Double-buffered LDS pipeline: next tile's global loads are issued, current
// tile's fragments are batch-read into registers (so WMMAs only need partial
// dscnt waits), next tile is stored into the alternate LDS buffer, then the
// 8 WMMAs run while the following weight panel streams from HBM.
// ---------------------------------------------------------------------------
__global__ __launch_bounds__(512)
void moe_gemm(const unsigned short* __restrict__ bufA, const float* __restrict__ we,
              float* __restrict__ yb) {
    __shared__ __align__(16) unsigned short As[2][C_][KTILE];     // 2 x 16 KB
    __shared__ __align__(16) unsigned short Bt[2][NTILE][KTILE];  // 2 x  8 KB (transposed)

    const int e     = blockIdx.y;
    const int nbase = blockIdx.x * NTILE;
    const int tid   = threadIdx.x;
    const int lane  = tid & 31;
    const int wv    = tid >> 5;                // 0..15
    const int hi    = lane >> 4;               // half-wave select

    const float*          W = we + (size_t)e * D_ * D_;
    const unsigned short* A = bufA + (size_t)e * C_ * D_;

    // per-thread staging coordinates
    const int am   = tid >> 1;                 // A row 0..255
    const int ah   = tid & 1;                  // which 32B half of the 64B row
    const int bk   = tid >> 4;                 // B k-row 0..31
    const int bn   = (tid & 15) * 8;           // B col block (8 cols)
    const unsigned short* agp = A + (size_t)am * D_ + ah * 16;  // +k0 later
    const float*          bgp = W + (size_t)bk * D_ + nbase + bn;

    v8f acc[8];
    const v8f vzero = {0.f, 0.f, 0.f, 0.f, 0.f, 0.f, 0.f, 0.f};
#pragma unroll
    for (int i = 0; i < 8; ++i) acc[i] = vzero;

    // ---- prologue: global-load tile 0 and store into LDS buffer 0 ----
    uint4  a0 = ((const uint4*)agp)[0];
    uint4  a1 = ((const uint4*)agp)[1];
    float4 f0 = ((const float4*)bgp)[0];
    float4 f1 = ((const float4*)bgp)[1];
    {
        uint4* dst = (uint4*)(&As[0][am][ah * 16]);
        dst[0] = a0; dst[1] = a1;
        Bt[0][bn + 0][bk] = f2bf(f0.x); Bt[0][bn + 1][bk] = f2bf(f0.y);
        Bt[0][bn + 2][bk] = f2bf(f0.z); Bt[0][bn + 3][bk] = f2bf(f0.w);
        Bt[0][bn + 4][bk] = f2bf(f1.x); Bt[0][bn + 5][bk] = f2bf(f1.y);
        Bt[0][bn + 6][bk] = f2bf(f1.z); Bt[0][bn + 7][bk] = f2bf(f1.w);
    }
    __syncthreads();

    int p = 0;
    for (int k0 = 0; k0 < D_; k0 += KTILE) {
        const bool hasNext = (k0 + KTILE) < D_;
        // ---- issue next tile's global loads (in flight under the LDS fragment reads) ----
        if (hasNext) {
            const unsigned short* ag = agp + (k0 + KTILE);
            const float*          bg = bgp + (size_t)(k0 + KTILE) * D_;
            a0 = ((const uint4*)ag)[0];
            a1 = ((const uint4*)ag)[1];
            f0 = ((const float4*)bg)[0];
            f1 = ((const float4*)bg)[1];
            if (k0 + 2 * KTILE < D_)   // cover HBM latency 2 panels ahead
                __builtin_prefetch(bgp + (size_t)(k0 + 2 * KTILE) * D_, 0, 1);
        }
        // ---- batch-read all fragments of the current tile into registers ----
        FragBF af;
        {
            const uint4* row = (const uint4*)(&As[p][wv * 16 + (lane & 15)][0]);
            af.q[0] = row[hi];        // lanes 0-15: K=0-7 / lanes 16-31: K=8-15
            af.q[1] = row[2 + hi];    // K=16-23 / K=24-31
        }
        FragBF bfg[8];
#pragma unroll
        for (int ni = 0; ni < 8; ++ni) {
            const uint4* row = (const uint4*)(&Bt[p][ni * 16 + (lane & 15)][0]);
            bfg[ni].q[0] = row[hi];       // lanes 0-15: K=0-7 / lanes 16-31: K=8-15
            bfg[ni].q[1] = row[2 + hi];   // K=16-23 / K=24-31
        }
        // ---- store next tile into the alternate LDS buffer ----
        if (hasNext) {
            const int q = p ^ 1;
            uint4* dst = (uint4*)(&As[q][am][ah * 16]);
            dst[0] = a0; dst[1] = a1;
            Bt[q][bn + 0][bk] = f2bf(f0.x); Bt[q][bn + 1][bk] = f2bf(f0.y);
            Bt[q][bn + 2][bk] = f2bf(f0.z); Bt[q][bn + 3][bk] = f2bf(f0.w);
            Bt[q][bn + 4][bk] = f2bf(f1.x); Bt[q][bn + 5][bk] = f2bf(f1.y);
            Bt[q][bn + 6][bk] = f2bf(f1.z); Bt[q][bn + 7][bk] = f2bf(f1.w);
        }
        __syncthreads();   // next iter may read the alternate buffer
        // ---- 8 WMMAs on registers only ----
#pragma unroll
        for (int ni = 0; ni < 8; ++ni)
            acc[ni] = __builtin_amdgcn_wmma_f32_16x16x32_bf16(
                false, af.v, false, bfg[ni].v, (short)0, acc[ni], false, false);
        p ^= 1;
    }

    // ---- write back: VGPR r of a 16x16 f32 tile holds M = r + 8*hi, N = lane&15
    const int mrow0 = wv * 16 + hi * 8;
    const int ncol  = nbase + (lane & 15);
#pragma unroll
    for (int ni = 0; ni < 8; ++ni) {
        float* out = yb + (size_t)(e * C_ + mrow0) * D_ + ncol + ni * 16;
#pragma unroll
        for (int r = 0; r < 8; ++r)
            out[(size_t)r * D_] = acc[ni][r];
    }
}

// ---------------------------------------------------------------------------
// Kernel 4: combine — gather both expert outputs, weight by normalized gates.
// ---------------------------------------------------------------------------
__global__ void moe_combine(const float* __restrict__ yb,
                            const int* __restrict__ e1a, const int* __restrict__ e2a,
                            const int* __restrict__ s1a, const int* __restrict__ s2a,
                            const float* __restrict__ g1a, const float* __restrict__ g2a,
                            float* __restrict__ out) {
    int t = blockIdx.x;
    int i = threadIdx.x;                       // 4 floats per thread
    int e1 = e1a[t], s1 = s1a[t], e2 = e2a[t], s2 = s2a[t];
    float g1 = g1a[t], g2 = g2a[t];
    float4 r = make_float4(0.f, 0.f, 0.f, 0.f);
    if (s1 < C_) {
        float4 a = ((const float4*)(yb + (size_t)(e1 * C_ + s1) * D_))[i];
        r.x += g1 * a.x; r.y += g1 * a.y; r.z += g1 * a.z; r.w += g1 * a.w;
    }
    if (s2 < C_) {
        float4 b = ((const float4*)(yb + (size_t)(e2 * C_ + s2) * D_))[i];
        r.x += g2 * b.x; r.y += g2 * b.y; r.z += g2 * b.z; r.w += g2 * b.w;
    }
    ((float4*)(out + (size_t)t * D_))[i] = r;
}

// ---------------------------------------------------------------------------
// Launch: zero -> gate -> dispatch -> grouped GEMM -> combine (all on stream).
// Workspace layout (~202 MB): bf16 dispatch buf (64MB) | f32 yb (128MB) | meta.
// ---------------------------------------------------------------------------
extern "C" void kernel_launch(void* const* d_in, const int* in_sizes, int n_in,
                              void* d_out, int out_size, void* d_ws, size_t ws_size,
                              hipStream_t stream) {
    (void)in_sizes; (void)n_in; (void)out_size; (void)ws_size;
    const float* x  = (const float*)d_in[0];   // [4,2048,1024]
    const float* wg = (const float*)d_in[1];   // [1024,128]
    const float* we = (const float*)d_in[2];   // [128,1024,1024]
    float* out = (float*)d_out;

    char* ws = (char*)d_ws;
    unsigned short* bufA = (unsigned short*)ws;                       // E*C*D bf16
    float* yb = (float*)(ws + (size_t)E_ * C_ * D_ * 2);              // E*C*D f32
    char* meta = ws + (size_t)E_ * C_ * D_ * 2 + (size_t)E_ * C_ * D_ * 4;
    int*   cnt = (int*)meta;
    int*   e1a = cnt + E_;
    int*   e2a = e1a + T_;
    int*   s1a = e2a + T_;
    int*   s2a = s1a + T_;
    float* g1a = (float*)(s2a + T_);
    float* g2a = g1a + T_;

    moe_zero<<<(E_ * C_ * D_ / 8 + 255) / 256, 256, 0, stream>>>((uint4*)bufA, cnt);
    moe_gate<<<T_, 128, 0, stream>>>(x, wg, cnt, e1a, e2a, s1a, s2a, g1a, g2a);
    moe_dispatch<<<T_, 256, 0, stream>>>(x, e1a, e2a, s1a, s2a, bufA);
    moe_gemm<<<dim3(D_ / NTILE, E_), 512, 0, stream>>>(bufA, we, yb);
    moe_combine<<<T_, 256, 0, stream>>>(yb, e1a, e2a, s1a, s2a, g1a, g2a, out);
}
